// Multi_Head_Self_Attention_58119497450415
// MI455X (gfx1250) — matrix operand order repaired and checked by reading, not compiled, not run on hardware
//
#include <hip/hip_runtime.h>
#include <math.h>
#include <stdint.h>

#define B_    4
#define M_    2048
#define H_    1024
#define NH_   16
#define D_    64
#define NROWS (B_ * M_)   // 8192

typedef __bf16 bf16;
typedef __attribute__((ext_vector_type(16))) __bf16 bf16x16;
typedef __attribute__((ext_vector_type(8)))  __bf16 bf16x8;
typedef __attribute__((ext_vector_type(8)))  float  f32x8;

union AFrag { bf16x16 v; bf16x8 h[2]; };

// gfx1250 async global->LDS copy (ASYNCcnt-tracked) via inline asm — the
// ROCm 7.2 builtin has a non-standard prototype, so we bypass it entirely.
#if defined(__gfx1250__)
#define HAVE_ASYNC_LDS 1
#else
#define HAVE_ASYNC_LDS 0
#endif

__device__ __forceinline__ void cp_g2s_b128(bf16* lds, const bf16* g) {
#if HAVE_ASYNC_LDS
  uint32_t l = (uint32_t)(uintptr_t)lds;        // LDS byte offset (addr[31:0])
  uint64_t ga = (uint64_t)(uintptr_t)g;         // 64-bit global address
  asm volatile("global_load_async_to_lds_b128 %0, %1, off"
               :
               : "v"(l), "v"(ga)
               : "memory");
#else
  *(bf16x8*)lds = *(const bf16x8*)g;
#endif
}

__device__ __forceinline__ void wait_stage() {
#if HAVE_ASYNC_LDS
  asm volatile("s_wait_asynccnt 0" ::: "memory");
#endif
  __syncthreads();
}

// ---------------------------------------------------------------------------
// WMMA helper: D = A(16x32 bf16) * B(32x16 bf16) + C(16x16 f32)
// ---------------------------------------------------------------------------
__device__ __forceinline__ f32x8 wmma_bf16(bf16x16 a, bf16x16 b, f32x8 c) {
  return __builtin_amdgcn_wmma_f32_16x16x32_bf16(
      false, a, false, b, (short)0, c, false, false);
}

// A fragment (16x32, MxK) from a row-major bf16 matrix [row][k], ld elems.
__device__ __forceinline__ bf16x16 load_a_frag(const bf16* base, int ld,
                                               int row0, int k0, int lane) {
  AFrag f;
  int r  = row0 + (lane & 15);
  int kb = k0 + ((lane >> 4) << 3);
  const bf16* p = base + (long)r * ld + kb;
  f.h[0] = *(const bf16x8*)(p);
  f.h[1] = *(const bf16x8*)(p + 16);
  return f.v;
}

// B fragment (32x16, KxN) from an N-major matrix T[n][k], ld elems.
__device__ __forceinline__ bf16x16 load_b_frag(const bf16* baseT, int ld,
                                               int n0, int k0, int lane) {
  AFrag f;
  int n  = n0 + (lane & 15);
  int kb = k0 + ((lane >> 4) << 3);
  const bf16* p = baseT + (long)n * ld + kb;
  f.h[0] = *(const bf16x8*)(p);
  f.h[1] = *(const bf16x8*)(p + 16);
  return f.v;
}

// ---------------------------------------------------------------------------
// Elementwise converts
// ---------------------------------------------------------------------------
__global__ void mhsa_cvt_bf16(const float* __restrict__ src,
                              bf16* __restrict__ dst, int n) {
  int i = blockIdx.x * blockDim.x + threadIdx.x;
  if (i < n) dst[i] = (bf16)src[i];
}

// W [k][n] (HxH f32) -> Wt [n][k] bf16
__global__ void mhsa_transpose_bf16(const float* __restrict__ W,
                                    bf16* __restrict__ Wt) {
  int i = blockIdx.x * blockDim.x + threadIdx.x;
  int k = i >> 10;
  int n = i & (H_ - 1);
  Wt[(long)n * H_ + k] = (bf16)W[(long)k * H_ + n];
}

// ---------------------------------------------------------------------------
// QKV projection GEMM. Block = 128 threads (4 waves), block tile 64x128,
// wave tile 32x64 (A frags reused across 4 B tiles).
// vmode 0: dst[bh][m][d] ; vmode 1: dst[bh][d][m]
// ---------------------------------------------------------------------------
__global__ __launch_bounds__(128)
void mhsa_gemm_qkv(const bf16* __restrict__ X, const bf16* __restrict__ Wt,
                   const float* __restrict__ bias, bf16* __restrict__ dst,
                   float scale, int vmode) {
  int lane = threadIdx.x & 31;
  int wave = threadIdx.x >> 5;
  int row0 = blockIdx.x * 64 + (wave >> 1) * 32;
  int col0 = blockIdx.y * 128 + (wave & 1) * 64;

  f32x8 c[2][4] = {};
  for (int k0 = 0; k0 < H_; k0 += 32) {
    bf16x16 a0 = load_a_frag(X, H_, row0,      k0, lane);
    bf16x16 a1 = load_a_frag(X, H_, row0 + 16, k0, lane);
#pragma unroll
    for (int tj = 0; tj < 4; ++tj) {
      bf16x16 b = load_b_frag(Wt, H_, col0 + tj * 16, k0, lane);
      c[0][tj] = wmma_bf16(a0, b, c[0][tj]);
      c[1][tj] = wmma_bf16(a1, b, c[1][tj]);
    }
  }

  int nloc = lane & 15;
  int moff = (lane >> 4) * 8;
#pragma unroll
  for (int ti = 0; ti < 2; ++ti)
#pragma unroll
    for (int tj = 0; tj < 4; ++tj)
#pragma unroll
      for (int r = 0; r < 8; ++r) {
        int grow = row0 + ti * 16 + moff + r;
        int gcol = col0 + tj * 16 + nloc;
        float v = (c[ti][tj][r] + bias[gcol]) * scale;
        int b = grow >> 11;
        int m = grow & (M_ - 1);
        int h = gcol >> 6;
        int d = gcol & (D_ - 1);
        long off = (vmode == 0)
                       ? (((long)(b * NH_ + h) * M_ + m) * D_ + d)
                       : (((long)(b * NH_ + h) * D_ + d) * M_ + m);
        dst[off] = (bf16)v;
      }
}

// ---------------------------------------------------------------------------
// Flash attention per (b,h). Block = 128 threads (4 waves) = 64 query rows.
// K/V chunks (64 keys) staged into LDS once per block via async copies,
// double-buffered: prefetch chunk i+1 while computing chunk i.
// ---------------------------------------------------------------------------
__global__ __launch_bounds__(128)
void mhsa_flash_attn(const bf16* __restrict__ Qh, const bf16* __restrict__ Kh,
                     const bf16* __restrict__ Vt, const int* __restrict__ masks,
                     bf16* __restrict__ ctx) {
  __shared__ __attribute__((aligned(16))) bf16 Ksh[2][64][72];
  __shared__ __attribute__((aligned(16))) bf16 Vsh[2][64][72];
  __shared__ __attribute__((aligned(16))) bf16 Plds[4][16][72];

  int tid  = threadIdx.x;
  int lane = tid & 31;
  int wave = tid >> 5;
  int bh = blockIdx.y;
  int b  = bh >> 4;
  int h  = bh & 15;
  int r0 = blockIdx.x * 64 + wave * 16;

  const bf16* Q = Qh + (long)bh * M_ * D_;
  const bf16* K = Kh + (long)bh * M_ * D_;   // [m][d]
  const bf16* V = Vt + (long)bh * D_ * M_;   // [d][m]
  const int* mask = masks + b * M_;

  // stage one 64-key chunk of K and V into LDS buffer `buf`
  auto stage = [&](int n0, int buf) {
#pragma unroll
    for (int it = 0; it < 4; ++it) {
      int idx = it * 128 + tid;      // 0..511, each covers 8 bf16 (16B)
      int rr  = idx >> 3;            // 0..63
      int cc  = (idx & 7) * 8;       // 0..56
      cp_g2s_b128(&Ksh[buf][rr][cc], K + (long)(n0 + rr) * D_ + cc); // key rr, d cc
      cp_g2s_b128(&Vsh[buf][rr][cc], V + (long)rr * M_ + n0 + cc);   // d rr, key cc
    }
  };

  bf16x16 aq0 = load_a_frag(Q, D_, r0, 0,  lane);
  bf16x16 aq1 = load_a_frag(Q, D_, r0, 32, lane);

  f32x8 o[4] = {};
  float mi[8], li[8];
#pragma unroll
  for (int r = 0; r < 8; ++r) { mi[r] = -1e30f; li[r] = 0.f; }

  int nloc = lane & 15;
  int moff = (lane >> 4) * 8;

  stage(0, 0);
  wait_stage();

  for (int ch = 0; ch < M_ / 64; ++ch) {
    int n0  = ch * 64;
    int buf = ch & 1;
    if (ch + 1 < M_ / 64) stage(n0 + 64, buf ^ 1);

    // --- S = Q K^T : B frags from LDS (K chunk shared by all 4 waves) ---
    f32x8 s[4] = {};
#pragma unroll
    for (int nt = 0; nt < 4; ++nt) {
      bf16x16 bk0 = load_b_frag(&Ksh[buf][0][0], 72, nt * 16, 0,  lane);
      bf16x16 bk1 = load_b_frag(&Ksh[buf][0][0], 72, nt * 16, 32, lane);
      s[nt] = wmma_bf16(aq0, bk0, s[nt]);
      s[nt] = wmma_bf16(aq1, bk1, s[nt]);
    }
    // --- mask ---
#pragma unroll
    for (int nt = 0; nt < 4; ++nt) {
      if (mask[n0 + nt * 16 + nloc] == 0) {
#pragma unroll
        for (int r = 0; r < 8; ++r) s[nt][r] = -1e12f;
      }
    }
    // --- online softmax ---
    float mnew[8], alpha[8];
#pragma unroll
    for (int r = 0; r < 8; ++r) {
      float v = fmaxf(fmaxf(s[0][r], s[1][r]), fmaxf(s[2][r], s[3][r]));
      v = fmaxf(v, __shfl_xor(v, 1, 32));
      v = fmaxf(v, __shfl_xor(v, 2, 32));
      v = fmaxf(v, __shfl_xor(v, 4, 32));
      v = fmaxf(v, __shfl_xor(v, 8, 32));
      mnew[r]  = fmaxf(mi[r], v);
      alpha[r] = __expf(mi[r] - mnew[r]);
    }
    float rsum[8] = {0.f, 0.f, 0.f, 0.f, 0.f, 0.f, 0.f, 0.f};
#pragma unroll
    for (int nt = 0; nt < 4; ++nt)
#pragma unroll
      for (int r = 0; r < 8; ++r) {
        float p = __expf(s[nt][r] - mnew[r]);
        rsum[r] += p;
        Plds[wave][moff + r][nt * 16 + nloc] = (bf16)p;
      }
#pragma unroll
    for (int r = 0; r < 8; ++r) {
      float v = rsum[r];
      v += __shfl_xor(v, 1, 32);
      v += __shfl_xor(v, 2, 32);
      v += __shfl_xor(v, 4, 32);
      v += __shfl_xor(v, 8, 32);
      li[r] = li[r] * alpha[r] + v;
      mi[r] = mnew[r];
    }
#pragma unroll
    for (int dt = 0; dt < 4; ++dt)
#pragma unroll
      for (int r = 0; r < 8; ++r) o[dt][r] *= alpha[r];

    asm volatile("" ::: "memory");   // same-wave DS ops are in-order

    // --- O += P * V : A from Plds, B from Vsh ---
#pragma unroll
    for (int ks = 0; ks < 2; ++ks) {
      AFrag ap;
      {
        int rr = lane & 15;
        int kb = ks * 32 + ((lane >> 4) << 3);
        const bf16* p = &Plds[wave][rr][kb];
        ap.h[0] = *(const bf16x8*)(p);
        ap.h[1] = *(const bf16x8*)(p + 16);
      }
#pragma unroll
      for (int dt = 0; dt < 4; ++dt) {
        bf16x16 bv = load_b_frag(&Vsh[buf][0][0], 72, dt * 16, ks * 32, lane);
        o[dt] = wmma_bf16(ap.v, bv, o[dt]);
      }
    }
    // prefetch done + everyone finished reading `buf` before it is reused
    wait_stage();
  }

  // --- writeout: ctx[b][m][h*64+d] bf16 ---
#pragma unroll
  for (int dt = 0; dt < 4; ++dt)
#pragma unroll
    for (int r = 0; r < 8; ++r) {
      int m = r0 + moff + r;
      float v = o[dt][r] / li[r];
      ctx[((long)(b * M_ + m)) * H_ + h * 64 + dt * 16 + nloc] = (bf16)v;
    }
}

// ---------------------------------------------------------------------------
// Output projection + mish, f32 out. Same tiling as QKV GEMM.
// ---------------------------------------------------------------------------
__global__ __launch_bounds__(128)
void mhsa_gemm_out(const bf16* __restrict__ Cx, const bf16* __restrict__ Wt,
                   const float* __restrict__ bias, float* __restrict__ out) {
  int lane = threadIdx.x & 31;
  int wave = threadIdx.x >> 5;
  int row0 = blockIdx.x * 64 + (wave >> 1) * 32;
  int col0 = blockIdx.y * 128 + (wave & 1) * 64;

  f32x8 c[2][4] = {};
  for (int k0 = 0; k0 < H_; k0 += 32) {
    bf16x16 a0 = load_a_frag(Cx, H_, row0,      k0, lane);
    bf16x16 a1 = load_a_frag(Cx, H_, row0 + 16, k0, lane);
#pragma unroll
    for (int tj = 0; tj < 4; ++tj) {
      bf16x16 b = load_b_frag(Wt, H_, col0 + tj * 16, k0, lane);
      c[0][tj] = wmma_bf16(a0, b, c[0][tj]);
      c[1][tj] = wmma_bf16(a1, b, c[1][tj]);
    }
  }

  int nloc = lane & 15;
  int moff = (lane >> 4) * 8;
#pragma unroll
  for (int ti = 0; ti < 2; ++ti)
#pragma unroll
    for (int tj = 0; tj < 4; ++tj)
#pragma unroll
      for (int r = 0; r < 8; ++r) {
        int grow = row0 + ti * 16 + moff + r;
        int gcol = col0 + tj * 16 + nloc;
        float v  = c[ti][tj][r] + bias[gcol];
        float sp = (v > 20.f) ? v : log1pf(__expf(v));   // softplus
        out[(long)grow * H_ + gcol] = v * tanhf(sp);     // mish
      }
}

// ---------------------------------------------------------------------------
// Launch
// ---------------------------------------------------------------------------
extern "C" void kernel_launch(void* const* d_in, const int* in_sizes, int n_in,
                              void* d_out, int out_size, void* d_ws, size_t ws_size,
                              hipStream_t stream) {
  const float* X  = (const float*)d_in[0];
  const int*   mk = (const int*)d_in[1];
  const float* Wq = (const float*)d_in[2];
  const float* bq = (const float*)d_in[3];
  const float* Wk = (const float*)d_in[4];
  const float* bk = (const float*)d_in[5];
  const float* Wv = (const float*)d_in[6];
  const float* bv = (const float*)d_in[7];
  const float* Wo = (const float*)d_in[8];
  const float* bo = (const float*)d_in[9];
  float* out = (float*)d_out;

  char* ws = (char*)d_ws;
  bf16* Xbf  = (bf16*)ws; ws += (size_t)NROWS * H_ * 2;
  bf16* Wqt  = (bf16*)ws; ws += (size_t)H_ * H_ * 2;
  bf16* Wkt  = (bf16*)ws; ws += (size_t)H_ * H_ * 2;
  bf16* Wvt  = (bf16*)ws; ws += (size_t)H_ * H_ * 2;
  bf16* Wot  = (bf16*)ws; ws += (size_t)H_ * H_ * 2;
  bf16* Qh   = (bf16*)ws; ws += (size_t)NROWS * H_ * 2;   // [bh][m][d]
  bf16* Kh   = (bf16*)ws; ws += (size_t)NROWS * H_ * 2;   // [bh][m][d]
  bf16* Vt   = (bf16*)ws; ws += (size_t)NROWS * H_ * 2;   // [bh][d][m]
  bf16* ctxb = (bf16*)ws; ws += (size_t)NROWS * H_ * 2;   // [b][m][H]

  int nX = NROWS * H_;
  mhsa_cvt_bf16<<<(nX + 255) / 256, 256, 0, stream>>>(X, Xbf, nX);
  int nW = H_ * H_;
  mhsa_transpose_bf16<<<(nW + 255) / 256, 256, 0, stream>>>(Wq, Wqt);
  mhsa_transpose_bf16<<<(nW + 255) / 256, 256, 0, stream>>>(Wk, Wkt);
  mhsa_transpose_bf16<<<(nW + 255) / 256, 256, 0, stream>>>(Wv, Wvt);
  mhsa_transpose_bf16<<<(nW + 255) / 256, 256, 0, stream>>>(Wo, Wot);

  dim3 gproj(NROWS / 64, H_ / 128);
  mhsa_gemm_qkv<<<gproj, 128, 0, stream>>>(Xbf, Wqt, bq, Qh, 0.125f, 0);
  mhsa_gemm_qkv<<<gproj, 128, 0, stream>>>(Xbf, Wkt, bk, Kh, 1.0f, 0);
  mhsa_gemm_qkv<<<gproj, 128, 0, stream>>>(Xbf, Wvt, bv, Vt, 1.0f, 1);

  dim3 gattn(M_ / 64, B_ * NH_);
  mhsa_flash_attn<<<gattn, 128, 0, stream>>>(Qh, Kh, Vt, mk, ctxb);

  mhsa_gemm_out<<<gproj, 128, 0, stream>>>(ctxb, Wot, bo, out);
}
